// EdgeAtt_4037269259072
// MI455X (gfx1250) — matrix operand order repaired and checked by reading, not compiled, not run on hardware
//
#include <hip/hip_runtime.h>

#define NB   512
#define LL   110
#define GG   512
#define LPAD 128
#define WPWF 10

typedef __attribute__((ext_vector_type(8)))  float  v8f;
typedef __attribute__((ext_vector_type(8)))  __bf16 v8bf;
typedef __attribute__((ext_vector_type(16))) __bf16 v16bf;

__device__ __forceinline__ unsigned short f32_to_bf16_bits(float f) {
  unsigned int u = __float_as_uint(f);
  u += 0x7FFFu + ((u >> 16) & 1u);      // round-to-nearest-even
  return (unsigned short)(u >> 16);
}

// Build a 16-element bf16 A fragment from two adjacent 16-byte chunks
// (contiguous K order: lane half holds K {0..15} / {16..31}).
__device__ __forceinline__ v16bf load_a_frag(const unsigned short* p) {
  v8bf a0 = *(const v8bf*)(p);
  v8bf a1 = *(const v8bf*)(p + 8);
  v16bf a;
#pragma unroll
  for (int i = 0; i < 8; ++i) { a[i] = a0[i]; a[i + 8] = a1[i]; }
  return a;
}

// ---- prep: W (f32, 512x512) -> bf16 in workspace (L2-resident thereafter) ----
__global__ void convert_weight_bf16(const float* __restrict__ w,
                                    unsigned short* __restrict__ wbf) {
  int i = blockIdx.x * blockDim.x + threadIdx.x;
  if (i < GG * GG) wbf[i] = f32_to_bf16_bits(w[i]);
}

// ---- fused: wn = nf_b @ W^T ; banded scores ; masked softmax ; write probs ----
__global__ __launch_bounds__(256)
void edge_att_kernel(const float* __restrict__ nf_g,
                     const int* __restrict__ text_len,
                     const unsigned short* __restrict__ wbf,
                     float* __restrict__ out) {
  extern __shared__ unsigned char smem[];
  unsigned short* sA  = (unsigned short*)smem;        // 128 x 512 bf16 (nf_b, zero-padded)
  unsigned short* sWN = sA + LPAD * GG;               // 128 x 512 bf16 (wn_b)
  float* sSC = (float*)(smem + 2u * LPAD * GG * sizeof(unsigned short)); // 7*16*48 f32

  const int b    = blockIdx.x;
  const int tid  = threadIdx.x;
  const int wid  = tid >> 5;    // wave id, 8 waves
  const int lane = tid & 31;
  const int ln16 = lane & 15;
  const int hf   = lane >> 4;   // lane half (WMMA fragment half)

  const float* nf = nf_g + (size_t)b * LL * GG;
  float* outb = out + (size_t)b * LL * LL;

  // Phase A0: zero this batch's 110x110 output slab (band is filled later)
  for (int i = tid; i < LL * LL; i += 256) outb[i] = 0.0f;

  // Phase A1: stream nf_b -> LDS as bf16, zero-pad rows [110,128)
  for (int i = tid; i < LPAD * GG / 4; i += 256) {
    int idx = i * 4;
    int row = idx >> 9;                 // /512
    unsigned int p0 = 0u, p1 = 0u;
    if (row < LL) {
      float4 v = *(const float4*)(nf + idx);
      p0 = (unsigned int)f32_to_bf16_bits(v.x) | ((unsigned int)f32_to_bf16_bits(v.y) << 16);
      p1 = (unsigned int)f32_to_bf16_bits(v.z) | ((unsigned int)f32_to_bf16_bits(v.w) << 16);
    }
    *(uint2*)(sA + idx) = make_uint2(p0, p1);
  }
  __syncthreads();

  // Phase B: WN(128x512) = A(128x512) @ W^T(512x512), bf16 WMMA, K-depth 512.
  // 2x2 register blocking: each wave owns a 32x32 macro-tile -> each A/B
  // fragment feeds two WMMAs (halves LDS and L2 operand traffic), and the
  // four accumulator chains give independent XDL work.
  {
    const int ka = hf ? 16 : 0;   // A-frag K base per lane half
    const int kb = hf ? 16 : 0;   // B-frag K base per lane half
    for (int task = wid; task < (LPAD / 32) * (GG / 32); task += 8) {
      int mt = task >> 4;         // 0..3   (M base = mt*32)
      int nt = task & 15;         // 0..15  (N base = nt*32)
      v8f acc00 = {}, acc01 = {}, acc10 = {}, acc11 = {};
      const unsigned short* arow0 = sA  + (size_t)(mt * 32 + ln16) * GG;
      const unsigned short* arow1 = arow0 + 16 * GG;
      const unsigned short* brow0 = wbf + (size_t)(nt * 32 + ln16) * GG;
      const unsigned short* brow1 = brow0 + 16 * GG;
#pragma unroll 4
      for (int kk = 0; kk < GG / 32; ++kk) {
        int k0 = kk * 32;
        v16bf a0 = load_a_frag(arow0 + k0 + ka);
        v16bf a1 = load_a_frag(arow1 + k0 + ka);
        v16bf b0 = *(const v16bf*)(brow0 + k0 + kb);   // 16 contiguous bf16 of W row
        v16bf b1 = *(const v16bf*)(brow1 + k0 + kb);
        acc00 = __builtin_amdgcn_wmma_f32_16x16x32_bf16(false, a0, false, b0,
                                                        (short)0, acc00, false, false);
        acc01 = __builtin_amdgcn_wmma_f32_16x16x32_bf16(false, a0, false, b1,
                                                        (short)0, acc01, false, false);
        acc10 = __builtin_amdgcn_wmma_f32_16x16x32_bf16(false, a1, false, b0,
                                                        (short)0, acc10, false, false);
        acc11 = __builtin_amdgcn_wmma_f32_16x16x32_bf16(false, a1, false, b1,
                                                        (short)0, acc11, false, false);
      }
      // C/D layout: vgpr r -> (M = r + 8*hf, N = ln16); store bf16 to sWN
      int m0 = mt * 32 + hf * 8;
      int n0 = nt * 32 + ln16;
#pragma unroll
      for (int r = 0; r < 8; ++r) {
        sWN[(size_t)(m0 + r) * GG + n0]           = f32_to_bf16_bits(acc00[r]);
        sWN[(size_t)(m0 + r) * GG + n0 + 16]      = f32_to_bf16_bits(acc01[r]);
        sWN[(size_t)(m0 + 16 + r) * GG + n0]      = f32_to_bf16_bits(acc10[r]);
        sWN[(size_t)(m0 + 16 + r) * GG + n0 + 16] = f32_to_bf16_bits(acc11[r]);
      }
    }
  }
  __syncthreads();

  // Phase C: banded scores S = A_j @ WN_k^T (<=3 diagonal 16x16 tiles per
  // j-tile), masked softmax, band store. Waves 0..6 each own one j-tile.
  const int cur = text_len[b];
  if (wid < 7) {
    const int t = wid;
    if (t * 16 < cur) {
      const int kt0 = (t > 0) ? t - 1 : 0;
      const int nkt = ((t > 0) && (t < 6)) ? 3 : 2;   // wave-uniform
      const int ka = hf ? 16 : 0;
      const unsigned short* arow = sA + (size_t)(t * 16 + ln16) * GG;
      const unsigned short* wn0  = sWN + (size_t)(kt0 * 16 + ln16) * GG + (hf ? 16 : 0);
      v8f acc0 = {}, acc1 = {}, acc2 = {};
#pragma unroll 4
      for (int kk = 0; kk < GG / 32; ++kk) {
        int k0 = kk * 32;
        v16bf a = load_a_frag(arow + k0 + ka);
        v16bf b0 = *(const v16bf*)(wn0 + k0);
        acc0 = __builtin_amdgcn_wmma_f32_16x16x32_bf16(false, a, false, b0,
                                                       (short)0, acc0, false, false);
        v16bf b1 = *(const v16bf*)(wn0 + 16 * GG + k0);
        acc1 = __builtin_amdgcn_wmma_f32_16x16x32_bf16(false, a, false, b1,
                                                       (short)0, acc1, false, false);
        if (nkt > 2) {                       // uniform branch: EXEC stays all-1s
          v16bf b2 = *(const v16bf*)(wn0 + 32 * GG + k0);
          acc2 = __builtin_amdgcn_wmma_f32_16x16x32_bf16(false, a, false, b2,
                                                         (short)0, acc2, false, false);
        }
      }
      // spill accumulators (16 rows x 48 band cols, f32) to this wave's LDS scratch
      float* sc = sSC + t * 16 * 48;
      int mb = hf * 8;
#pragma unroll
      for (int r = 0; r < 8; ++r) {
        sc[(mb + r) * 48 + ln16]      = acc0[r];
        sc[(mb + r) * 48 + 16 + ln16] = acc1[r];
        if (nkt > 2) sc[(mb + r) * 48 + 32 + ln16] = acc2[r];
      }
      // masked softmax: lanes 0..15 each own one row j = 16t+lane
      if (lane < 16) {
        int j = t * 16 + lane;
        if (j < cur && j < LL) {
          int lo = j - WPWF; if (lo < 0) lo = 0;
          int hi = j + WPWF; if (hi > cur - 1) hi = cur - 1;
          const float* srow = sc + lane * 48 - kt0 * 16;   // index by global k
          float mx = -3.0e38f;
          for (int k = lo; k <= hi; ++k) mx = fmaxf(mx, srow[k]);
          float sum = 0.0f;
          for (int k = lo; k <= hi; ++k) sum += __expf(srow[k] - mx);
          float inv = 1.0f / sum;
          float* orow = outb + (size_t)j * LL;
          for (int k = lo; k <= hi; ++k) orow[k] = __expf(srow[k] - mx) * inv;
        }
      }
    }
  }
}

extern "C" void kernel_launch(void* const* d_in, const int* in_sizes, int n_in,
                              void* d_out, int out_size, void* d_ws, size_t ws_size,
                              hipStream_t stream) {
  (void)in_sizes; (void)n_in; (void)out_size; (void)ws_size;
  const float* nf = (const float*)d_in[0];
  const int*   tl = (const int*)d_in[1];
  const float* w  = (const float*)d_in[3];
  unsigned short* wbf = (unsigned short*)d_ws;       // 512 KB of workspace

  convert_weight_bf16<<<(GG * GG + 255) / 256, 256, 0, stream>>>(w, wbf);

  size_t smem = 2u * LPAD * GG * sizeof(unsigned short)   // sA + sWN (256 KB)
              + 7u * 16u * 48u * sizeof(float);           // score scratch (21 KB)
  hipFuncSetAttribute((const void*)edge_att_kernel,
                      hipFuncAttributeMaxDynamicSharedMemorySize, (int)smem);
  edge_att_kernel<<<NB, 256, smem, stream>>>(nf, tl, wbf, (float*)d_out);
}
